// HGTDetector_90443421319185
// MI455X (gfx1250) — matrix operand order repaired and checked by reading, not compiled, not run on hardware
//
#include <hip/hip_runtime.h>
#include <math.h>

// ---------------- problem constants ----------------
#define U_CNT   8000
#define T_CNT   8
#define NT_CNT  64000
#define E_DIM   768
#define ED_DIM  128
#define H_CNT   4
#define DH_DIM  192
#define NCHUNK  8
#define CH_U    1000      // users per semantic-consistency chunk
#define CH_T    8000      // tweets per chunk
#define NE_FF   128000    // follow / friend edge count (U*DEG)
#define NE_P    64000     // post / rev_post edge count (NT)

typedef __attribute__((ext_vector_type(16))) __bf16    bf16x16;
typedef __attribute__((ext_vector_type(2)))  __bf16    bf16x2;
typedef __attribute__((ext_vector_type(8)))  float     f32x8;
typedef __attribute__((ext_vector_type(8)))  unsigned  u32x8;

__device__ __forceinline__ float waveSum(float v) {
#pragma unroll
  for (int m = 16; m > 0; m >>= 1) v += __shfl_xor(v, m, 32);
  return v;
}
__device__ __forceinline__ float leakyf(float v) { return v >= 0.f ? v : 0.01f * v; }

// pack two f32 into a dword of two bf16 (lowers to v_cvt_pk_bf16_f32)
__device__ __forceinline__ unsigned pkbf(float a, float b) {
  bf16x2 t; t[0] = (__bf16)a; t[1] = (__bf16)b;
  return __builtin_bit_cast(unsigned, t);
}

// order-preserving f32 <-> u32 mapping for atomic max
__device__ __forceinline__ unsigned fenc(float f) {
  unsigned u = __float_as_uint(f);
  return (u & 0x80000000u) ? ~u : (u | 0x80000000u);
}
__device__ __forceinline__ float fdec(unsigned u) {
  u = (u & 0x80000000u) ? (u & 0x7fffffffu) : ~u;
  return __uint_as_float(u);
}

// ---------------------------------------------------------------------------
// WMMA GEMM:  C[M,N] = act( A' @ B + bias ),  BM = 64 fixed, 8 waves/block.
//   GATHER : A'[r] = A[gidx[r]]   (graph-edge gather, indices hoisted to LDS)
//   TRANSW : B = W (K,N) right-multiply;  else B = W^T with W stored (N,K)
//   AVEC   : A rows are 16B-aligned with lda%4==0 -> b128 staging loads
// f32 -> bf16 conversion happens while staging into LDS (packed pairs, uint),
// 80B LDS row stride so fragment gathers merge into aligned ds_load_b128.
// Requires M%64==0, N%BN==0 (grid exact), K%32==0 — true at every call site.
// ---------------------------------------------------------------------------
template <int BN, bool GATHER, bool TRANSW, bool AVEC>
__global__ __launch_bounds__(256) void wmma_gemm(
    const float* __restrict__ A, int lda, const int* __restrict__ gidx,
    const float* __restrict__ W, int ldw,
    const float* __restrict__ bias,
    float* __restrict__ C, int ldc, int K, int act)
{
  constexpr int BM = 64;
  constexpr int TM = BM / 16, TN = BN / 16, TILES = TM * TN, TPW = TILES / 8;
  static_assert(TILES % 8 == 0, "tiles must split across 8 waves");
  constexpr int LSTR = 20;                 // uints per LDS row (16 K-pairs + pad)
  __shared__ unsigned As[BM * LSTR];
  __shared__ unsigned Bs[BN * LSTR];
  __shared__ int rowIdx[BM];

  const int tid = threadIdx.x;
  const int wv  = tid >> 5;
  const int ln  = tid & 31;
  const int hv  = ln >> 4;                 // lane half (0/1)
  const int l16 = ln & 15;
  const int bm0 = blockIdx.x * BM;
  const int bn0 = blockIdx.y * BN;

  if (GATHER) {                            // K-invariant: load indices once
    if (tid < BM) rowIdx[tid] = gidx[bm0 + tid];
    __syncthreads();
  }

  f32x8 acc[TPW];
#pragma unroll
  for (int t = 0; t < TPW; t++)
#pragma unroll
    for (int r = 0; r < 8; r++) acc[t][r] = 0.0f;

  for (int k0 = 0; k0 < K; k0 += 32) {
    // ---- stage A (64 x 32 f32 -> bf16 pairs) ----
    if (AVEC) {
      const int r0 = tid >> 3, c4 = (tid & 7) * 4;
      const int ra0 = GATHER ? rowIdx[r0]      : bm0 + r0;
      const int ra1 = GATHER ? rowIdx[r0 + 32] : bm0 + r0 + 32;
      const float4 va0 = *(const float4*)(A + (size_t)((unsigned)ra0 * lda + k0 + c4));
      const float4 va1 = *(const float4*)(A + (size_t)((unsigned)ra1 * lda + k0 + c4));
      unsigned* d0 = &As[r0 * LSTR + (c4 >> 1)];
      unsigned* d1 = &As[(r0 + 32) * LSTR + (c4 >> 1)];
      d0[0] = pkbf(va0.x, va0.y); d0[1] = pkbf(va0.z, va0.w);
      d1[0] = pkbf(va1.x, va1.y); d1[1] = pkbf(va1.z, va1.w);
    } else {
#pragma unroll
      for (int u = 0; u < 4; u++) {        // 1024 uints / 256 threads
        int i = tid + u * 256;
        int r = i >> 4, kk = i & 15;
        int arow = GATHER ? rowIdx[r] : bm0 + r;
        const float* ap = A + (size_t)((unsigned)arow * lda + k0 + 2 * kk);
        As[r * LSTR + kk] = pkbf(ap[0], ap[1]);
      }
    }
    // ---- stage B (BN x 32) ----
    if (!TRANSW) {                         // W (N,K) row-major: b128 along K
      const int r0 = tid >> 3, c4 = (tid & 7) * 4;
#pragma unroll
      for (int u = 0; u < BN / 32; u++) {
        int r = r0 + u * 32;
        float4 w = *(const float4*)(W + (size_t)((unsigned)(bn0 + r) * ldw + k0 + c4));
        unsigned* d = &Bs[r * LSTR + (c4 >> 1)];
        d[0] = pkbf(w.x, w.y); d[1] = pkbf(w.z, w.w);
      }
    } else {                               // W (K,N): b128 along N, pack K-pairs
      const int total = 16 * (BN / 4);
      for (int i = tid; i < total; i += 256) {
        int kk = i / (BN / 4), n = (i % (BN / 4)) * 4;
        const float* w0 = W + (size_t)((unsigned)(k0 + 2 * kk) * ldw + bn0 + n);
        float4 r0 = *(const float4*)w0;
        float4 r1 = *(const float4*)(w0 + ldw);
        Bs[(n + 0) * LSTR + kk] = pkbf(r0.x, r1.x);
        Bs[(n + 1) * LSTR + kk] = pkbf(r0.y, r1.y);
        Bs[(n + 2) * LSTR + kk] = pkbf(r0.z, r1.z);
        Bs[(n + 3) * LSTR + kk] = pkbf(r0.w, r1.w);
      }
    }
    __syncthreads();

#pragma unroll
    for (int t = 0; t < TPW; t++) {
      const int tile = wv * TPW + t;
      const int tr = tile % TM, tc = tile / TM;
      const unsigned* ar = &As[(tr * 16 + l16) * LSTR];
      const unsigned* br = &Bs[(tc * 16 + l16) * LSTR];
      u32x8 au, bu;
#pragma unroll
      for (int v = 0; v < 8; v++) {
        // A pair index: ka/2 = (v/4)*8 + half*4 + (v%4)  (ISA 7.12.2 16-bit A)
        au[v] = ar[((v >> 2) << 3) + (hv << 2) + (v & 3)];
        // B pair index: kb/2 = (v/4)*8 + half*4 + (v%4)  (same k order as A)
        bu[v] = br[((v >> 2) << 3) + (hv << 2) + (v & 3)];
      }
      acc[t] = __builtin_amdgcn_wmma_f32_16x16x32_bf16(
          false, __builtin_bit_cast(bf16x16, au),
          false, __builtin_bit_cast(bf16x16, bu),
          (short)0, acc[t], false, false);
    }
    __syncthreads();
  }

#pragma unroll
  for (int t = 0; t < TPW; t++) {
    const int tile = wv * TPW + t;
    const int tr = tile % TM, tc = tile / TM;
    const int col = bn0 + tc * 16 + l16;
    const float bv = bias ? bias[col] : 0.0f;
#pragma unroll
    for (int r = 0; r < 8; r++) {
      int row = bm0 + tr * 16 + hv * 8 + r;   // C layout: lane half -> M 0-7 / 8-15
      float v = acc[t][r] + bv;
      if (act == 1) v = leakyf(v);
      C[(size_t)((unsigned)row) * ldc + col] = v;
    }
  }
}

// ---------------------------------------------------------------------------
// Per-user multi-head self-attention over T=8 tweets + consistency pooling.
// One block (256 thr) per user. qkv rows: [q(768) | k(768) | v(768)].
// ---------------------------------------------------------------------------
__global__ __launch_bounds__(256) void k_attn(
    const float* __restrict__ qkv, float* __restrict__ oheads,
    float* __restrict__ cons, int uBase)
{
  __shared__ float s[H_CNT][T_CNT][T_CNT];
  __shared__ float p[H_CNT][T_CNT][T_CNT];
  __shared__ float aw[T_CNT][T_CNT];

  const int uL = blockIdx.x;
  const float* base = qkv + (size_t)uL * T_CNT * (3 * E_DIM);
  const int t = threadIdx.x;

  {   // scores: exactly 4*8*8 = 256 threads, vectorized dot
    int h = t >> 6, i = (t >> 3) & 7, j = t & 7;
    const float4* qi = (const float4*)(base + i * (3 * E_DIM) + h * DH_DIM);
    const float4* kj = (const float4*)(base + j * (3 * E_DIM) + E_DIM + h * DH_DIM);
    float a = 0.f;
#pragma unroll 4
    for (int d = 0; d < DH_DIM / 4; d++) {
      float4 q4 = qi[d], k4 = kj[d];
      a += q4.x * k4.x + q4.y * k4.y + q4.z * k4.z + q4.w * k4.w;
    }
    s[h][i][j] = a * 0.07216878364870323f;   // 1/sqrt(192)
  }
  __syncthreads();
  if (t < 32) {   // row softmax
    int h = t >> 3, i = t & 7;
    float m = -1e30f;
    for (int j = 0; j < 8; j++) m = fmaxf(m, s[h][i][j]);
    float e[8], sum = 0.f;
    for (int j = 0; j < 8; j++) { e[j] = expf(s[h][i][j] - m); sum += e[j]; }
    for (int j = 0; j < 8; j++) p[h][i][j] = e[j] / sum;
  }
  __syncthreads();
  if (t < 64) {   // head-mean attention
    int i = t >> 3, j = t & 7;
    aw[i][j] = 0.25f * (p[0][i][j] + p[1][i][j] + p[2][i][j] + p[3][i][j]);
  }
  __syncthreads();
  if (t < 16) {   // 2x2 max-pool -> consistency vector
    int a = t >> 2, c = t & 3;
    float m = fmaxf(fmaxf(aw[2 * a][2 * c], aw[2 * a][2 * c + 1]),
                    fmaxf(aw[2 * a + 1][2 * c], aw[2 * a + 1][2 * c + 1]));
    cons[(size_t)(uBase + uL) * 16 + a * 4 + c] = m;
  }
  // o[i, h*DH+d] = sum_j p[h][i][j] * v[j, h*DH+d]
  for (int oi = t; oi < T_CNT * E_DIM; oi += 256) {
    int i = oi / E_DIM, col = oi % E_DIM, h = col / DH_DIM;
    float a = 0.f;
#pragma unroll
    for (int j = 0; j < 8; j++) a += p[h][i][j] * base[j * (3 * E_DIM) + 2 * E_DIM + col];
    oheads[(size_t)uL * T_CNT * E_DIM + oi] = a;
  }
}

// LayerNorm over D (multiple of 32, <=768); optional elementwise add input.
__global__ __launch_bounds__(256) void k_ln(
    const float* __restrict__ x, const float* __restrict__ addx,
    const float* __restrict__ g, const float* __restrict__ be,
    float* __restrict__ y, int rows, int D)
{
  int wv = threadIdx.x >> 5, ln = threadIdx.x & 31;
  int row = blockIdx.x * 8 + wv;
  if (row >= rows) return;
  float vals[24];
  int n = D >> 5;
  const float* xr = x + (size_t)row * D;
  const float* ar = addx ? addx + (size_t)row * D : nullptr;
  float s = 0.f;
  for (int i = 0; i < n; i++) {
    float v = xr[ln + 32 * i];
    if (ar) v += ar[ln + 32 * i];
    vals[i] = v; s += v;
  }
  float mean = waveSum(s) / (float)D;
  float vs = 0.f;
  for (int i = 0; i < n; i++) { float d = vals[i] - mean; vs += d * d; }
  float inv = rsqrtf(waveSum(vs) / (float)D + 1e-5f);
  float* yr = y + (size_t)row * D;
  for (int i = 0; i < n; i++)
    yr[ln + 32 * i] = (vals[i] - mean) * inv * g[ln + 32 * i] + be[ln + 32 * i];
}

// Property branch: out32 = leaky(ln32(z@W^T + b)). One wave per user, lane = out dim.
// If W==nullptr, reads precomputed linear from h (for the WMMA-computed 'des' branch).
__global__ __launch_bounds__(256) void k_branch(
    const float* __restrict__ z, int ldz, int zoff, int K,
    const float* __restrict__ W, const float* __restrict__ b,
    const float* __restrict__ g, const float* __restrict__ be,
    float* __restrict__ h, int hoff, int rows)
{
  int wv = threadIdx.x >> 5, ln = threadIdx.x & 31;
  int u = blockIdx.x * 8 + wv;
  if (u >= rows) return;
  float val;
  if (W) {
    float a = b[ln];
    const float* zr = z + (size_t)u * ldz + zoff;
    const float* wr = W + (size_t)ln * K;
    for (int k = 0; k < K; k++) a += zr[k] * wr[k];
    val = a;
  } else {
    val = h[(size_t)u * ED_DIM + hoff + ln];
  }
  float mean = waveSum(val) * (1.f / 32.f);
  float d = val - mean;
  float var = waveSum(d * d) * (1.f / 32.f);
  float y = d * rsqrtf(var + 1e-5f) * g[ln] + be[ln];
  h[(size_t)u * ED_DIM + hoff + ln] = leakyf(y);
}

// al[e] = dot(qd[dst[e]], ke[e]) * p / sqrt(128).  One wave per edge.
__global__ __launch_bounds__(256) void k_edge_score(
    const float* __restrict__ qd, const int* __restrict__ dst,
    const float* __restrict__ ke, const float* __restrict__ p,
    float* __restrict__ al, int E)
{
  int wv = threadIdx.x >> 5, ln = threadIdx.x & 31;
  int e = blockIdx.x * 8 + wv;
  if (e >= E) return;
  const float* q = qd + (size_t)dst[e] * ED_DIM;
  const float* k = ke + (size_t)e * ED_DIM;
  float s = 0.f;
  for (int c = ln; c < ED_DIM; c += 32) s += q[c] * k[c];
  s = waveSum(s);
  if (ln == 0) al[e] = s * p[0] * 0.08838834764831845f;  // 1/sqrt(128)
}

__global__ void k_amax(const float* __restrict__ al, const int* __restrict__ dst,
                       unsigned* __restrict__ amax, int E)
{
  int e = blockIdx.x * 256 + threadIdx.x;
  if (e >= E) return;
  atomicMax(&amax[dst[e]], fenc(al[e]));
}

__global__ void k_expden(const float* __restrict__ al, const int* __restrict__ dst,
                         const unsigned* __restrict__ amax,
                         float* __restrict__ ex, float* __restrict__ den, int E)
{
  int e = blockIdx.x * 256 + threadIdx.x;
  if (e >= E) return;
  float m = fdec(amax[dst[e]]);
  if (!(fabsf(m) < 1e38f)) m = 0.f;   // non-finite segment max -> 0 (ref semantics)
  float v = expf(al[e] - m);
  ex[e] = v;
  atomicAdd(&den[dst[e]], v);
}

__global__ void k_scatter(const float* __restrict__ ex, const float* __restrict__ den,
                          const int* __restrict__ dst, const float* __restrict__ ve,
                          float* __restrict__ agg, int E)
{
  size_t tid = (size_t)blockIdx.x * 256 + threadIdx.x;
  if (tid >= (size_t)E * ED_DIM) return;
  int e = (int)(tid >> 7), c = (int)(tid & 127);
  float w = ex[e] / (den[dst[e]] + 1e-16f);
  atomicAdd(&agg[(size_t)dst[e] * ED_DIM + c], w * ve[(size_t)e * ED_DIM + c]);
}

__global__ void k_gelu(float* __restrict__ x, size_t n)
{
  size_t i = (size_t)blockIdx.x * 256 + threadIdx.x;
  if (i >= n) return;
  float v = x[i];
  x[i] = 0.5f * v * (1.f + erff(v * 0.7071067811865476f));
}

__global__ void k_blend(const float* __restrict__ a, const float* __restrict__ x,
                        const float* __restrict__ skip, float* __restrict__ out, size_t n)
{
  size_t i = (size_t)blockIdx.x * 256 + threadIdx.x;
  if (i >= n) return;
  float sk = 1.f / (1.f + expf(-skip[0]));
  out[i] = sk * a[i] + (1.f - sk) * x[i];
}

__global__ void k_cls(const float* __restrict__ h, const float* __restrict__ W2,
                      const float* __restrict__ b2, float* __restrict__ out, int n)
{
  int u = blockIdx.x * 256 + threadIdx.x;
  if (u >= n) return;
  const float* hr = h + (size_t)u * ED_DIM;
  float l0 = b2[0], l1 = b2[1];
  for (int c = 0; c < ED_DIM; c++) { l0 += hr[c] * W2[c]; l1 += hr[c] * W2[ED_DIM + c]; }
  float m = fmaxf(l0, l1);
  float e0 = expf(l0 - m), e1 = expf(l1 - m);
  float s = e0 + e1;
  out[2 * u] = e0 / s;
  out[2 * u + 1] = e1 / s;
}

// ---------------------------------------------------------------------------
// Host side
// ---------------------------------------------------------------------------
// dense row-major A (16B aligned, lda%4==0) x W(N,K)^T
static inline void gemm_nn(hipStream_t st, const float* A, int lda,
                           const float* W, int ldw, const float* bias,
                           float* C, int ldc, int M, int N, int K, int act)
{
  dim3 g(M / 64, N / 64);
  wmma_gemm<64, false, false, true><<<g, 256, 0, st>>>(A, lda, nullptr, W, ldw, bias, C, ldc, K, act);
}
// des branch: misaligned A (user_x+9, lda 793), N = 32
static inline void gemm_des(hipStream_t st, const float* A, int lda,
                            const float* W, int ldw, const float* bias,
                            float* C, int ldc, int M, int K)
{
  dim3 g(M / 64, 1);
  wmma_gemm<32, false, false, false><<<g, 256, 0, st>>>(A, lda, nullptr, W, ldw, bias, C, ldc, K, 0);
}
// graph-edge transform: C[e] = A[gidx[e]] @ W, W stored (128,128) un-transposed
static inline void gemm_edge(hipStream_t st, const float* A, const int* gidx,
                             const float* W, float* C, int M)
{
  dim3 g(M / 64, ED_DIM / 64);
  wmma_gemm<64, true, true, true><<<g, 256, 0, st>>>(A, ED_DIM, gidx, W, ED_DIM, nullptr, C, ED_DIM, ED_DIM, 0);
}

struct NodeP { const float *Wk,*bk,*Wq,*bq,*Wv,*bv,*Wa,*ba,*skip; };
struct EdgeP { const float *a,*m,*p; };
struct HgtP  { NodeP nu, nt; EdgeP follow, frnd, post, rev; };

extern "C" void kernel_launch(void* const* d_in, const int* in_sizes, int n_in,
                              void* d_out, int out_size, void* d_ws, size_t ws_size,
                              hipStream_t stream)
{
  (void)in_sizes; (void)n_in; (void)out_size; (void)ws_size;
  auto F = [&](int i) { return (const float*)d_in[i]; };
  auto I = [&](int i) { return (const int*)d_in[i]; };

  // ---- inputs in setup_inputs() insertion order (recursively) ----
  const float* user_x  = F(0);      // (8000, 793)
  const float* tweet_x = F(1);      // (64000, 768)
  const int*   e_fol   = I(2);      // (2, 128000)
  const int*   e_fri   = I(3);
  const int*   e_pst   = I(4);      // (2, 64000)
  const int*   e_rev   = I(5);
  // pv branches: W,b,g,be each
  const float *pc_W=F(6), *pc_b=F(7), *pc_g=F(8), *pc_be=F(9);        // cat  K=4
  const float *pn_W=F(10),*pn_b=F(11),*pn_g=F(12),*pn_be=F(13);       // num  K=5
  const float *pd_W=F(14),*pd_b=F(15),*pd_g=F(16),*pd_be=F(17);       // des  K=768
  const float *po_W=F(18),*po_b=F(19),*po_g=F(20),*po_be=F(21);       // con  K=16
  const float *out_g=F(22),*out_be=F(23),*out_W=F(24),*out_b=F(25);   // pv.out (g,be,W,b)
  // sc
  const float *Win=F(26),*bin=F(27),*Wo=F(28),*bo=F(29);
  const float *sc_g=F(30),*sc_be=F(31),*Wu=F(32),*bu=F(33);
  // tw
  const float *twW=F(34),*twb=F(35);
  auto node = [&](int b){ return NodeP{F(b),F(b+1),F(b+2),F(b+3),F(b+4),F(b+5),F(b+6),F(b+7),F(b+8)}; };
  auto edge = [&](int b){ return EdgeP{F(b),F(b+1),F(b+2)}; };
  HgtP H1{node(36), node(45), edge(54), edge(57), edge(60), edge(63)};
  HgtP H2{node(66), node(75), edge(84), edge(87), edge(90), edge(93)};
  const float *cW1=F(96),*cb1=F(97),*cW2=F(98),*cb2=F(99);

  // ---- workspace bump allocator (f32 elements, 256B aligned) ----
  float* ws = (float*)d_ws;
  size_t off = 0;
  auto alloc = [&](size_t n) { float* p = ws + off; off += (n + 63) & ~(size_t)63; return p; };

  // persistent buffers
  float* cons = alloc((size_t)U_CNT * 16);
  float* xt   = alloc((size_t)NT_CNT * ED_DIM);
  float* xu   = alloc((size_t)U_CNT * ED_DIM);
  float* xu1  = alloc((size_t)U_CNT * ED_DIM);
  float* xt1  = alloc((size_t)NT_CNT * ED_DIM);
  float* xu2  = alloc((size_t)U_CNT * ED_DIM);
  float* xt2  = alloc((size_t)NT_CNT * ED_DIM);
  const size_t arena = off;

  // =========== Phase 1: semantic consistency (chunked over users) ===========
  off = arena;
  float* qkv = alloc((size_t)CH_T * 3 * E_DIM);
  float* oh  = alloc((size_t)CH_T * E_DIM);
  float* op  = alloc((size_t)CH_T * E_DIM);
  float* yb  = alloc((size_t)CH_T * E_DIM);
  float* tb  = alloc((size_t)CH_T * E_DIM);
  for (int ch = 0; ch < NCHUNK; ch++) {
    const float* xch = tweet_x + (size_t)ch * CH_T * E_DIM;
    gemm_nn(stream, xch, E_DIM, Win, E_DIM, bin, qkv, 3*E_DIM, CH_T, 3*E_DIM, E_DIM, 0);
    k_attn<<<CH_U, 256, 0, stream>>>(qkv, oh, cons, ch * CH_U);
    gemm_nn(stream, oh, E_DIM, Wo, E_DIM, bo, op, E_DIM, CH_T, E_DIM, E_DIM, 0);
    k_ln<<<CH_T / 8, 256, 0, stream>>>(xch, op, sc_g, sc_be, yb, CH_T, E_DIM);
    gemm_nn(stream, yb, E_DIM, Wu, E_DIM, bu, tb, E_DIM, CH_T, E_DIM, E_DIM, 1);
    gemm_nn(stream, tb, E_DIM, twW, E_DIM, twb,
            xt + (size_t)ch * CH_T * ED_DIM, ED_DIM, CH_T, ED_DIM, E_DIM, 1);
  }

  // =========== Phase 2: property vector for users ===========
  off = arena;
  float* hb = alloc((size_t)U_CNT * ED_DIM);
  float* h2 = alloc((size_t)U_CNT * ED_DIM);
  k_branch<<<U_CNT / 8, 256, 0, stream>>>(user_x, 793, 0, 4,  pc_W, pc_b, pc_g, pc_be, hb, 0,  U_CNT);
  k_branch<<<U_CNT / 8, 256, 0, stream>>>(user_x, 793, 4, 5,  pn_W, pn_b, pn_g, pn_be, hb, 32, U_CNT);
  gemm_des(stream, user_x + 9, 793, pd_W, E_DIM, pd_b, hb + 64, ED_DIM, U_CNT, E_DIM);
  k_branch<<<U_CNT / 8, 256, 0, stream>>>(nullptr, 0, 0, 0, nullptr, nullptr, pd_g, pd_be, hb, 64, U_CNT);
  k_branch<<<U_CNT / 8, 256, 0, stream>>>(cons, 16, 0, 16, po_W, po_b, po_g, po_be, hb, 96, U_CNT);
  k_ln<<<U_CNT / 8, 256, 0, stream>>>(hb, nullptr, out_g, out_be, h2, U_CNT, ED_DIM);
  gemm_nn(stream, h2, ED_DIM, out_W, ED_DIM, out_b, xu, ED_DIM, U_CNT, ED_DIM, ED_DIM, 1);

  // =========== Phase 3: two HGT layers ===========
  off = arena;
  float* kdu = alloc((size_t)U_CNT * ED_DIM);
  float* qdu = alloc((size_t)U_CNT * ED_DIM);
  float* vdu = alloc((size_t)U_CNT * ED_DIM);
  float* kdt = alloc((size_t)NT_CNT * ED_DIM);
  float* qdt = alloc((size_t)NT_CNT * ED_DIM);
  float* vdt = alloc((size_t)NT_CNT * ED_DIM);
  float* aggu = alloc((size_t)U_CNT * ED_DIM);
  float* aggt = alloc((size_t)NT_CNT * ED_DIM);
  float* ke = alloc((size_t)NE_FF * ED_DIM);
  float* ve = alloc((size_t)NE_FF * ED_DIM);
  float* al = alloc(NE_FF);
  float* ex = alloc(NE_FF);
  unsigned* amax = (unsigned*)alloc(NT_CNT);
  float* den = alloc(NT_CNT);

  auto runHgt = [&](const HgtP& P, const float* xui, const float* xti,
                    float* xuo, float* xto) {
    gemm_nn(stream, xui, ED_DIM, P.nu.Wk, ED_DIM, P.nu.bk, kdu, ED_DIM, U_CNT, ED_DIM, ED_DIM, 0);
    gemm_nn(stream, xui, ED_DIM, P.nu.Wq, ED_DIM, P.nu.bq, qdu, ED_DIM, U_CNT, ED_DIM, ED_DIM, 0);
    gemm_nn(stream, xui, ED_DIM, P.nu.Wv, ED_DIM, P.nu.bv, vdu, ED_DIM, U_CNT, ED_DIM, ED_DIM, 0);
    gemm_nn(stream, xti, ED_DIM, P.nt.Wk, ED_DIM, P.nt.bk, kdt, ED_DIM, NT_CNT, ED_DIM, ED_DIM, 0);
    gemm_nn(stream, xti, ED_DIM, P.nt.Wq, ED_DIM, P.nt.bq, qdt, ED_DIM, NT_CNT, ED_DIM, ED_DIM, 0);
    gemm_nn(stream, xti, ED_DIM, P.nt.Wv, ED_DIM, P.nt.bv, vdt, ED_DIM, NT_CNT, ED_DIM, ED_DIM, 0);
    hipMemsetAsync(aggu, 0, (size_t)U_CNT * ED_DIM * 4, stream);
    hipMemsetAsync(aggt, 0, (size_t)NT_CNT * ED_DIM * 4, stream);

    auto rel = [&](const EdgeP& pe, const float* kds, const float* vds, const float* qdd,
                   const int* src, const int* dst, int E, int Nd, float* aggd) {
      gemm_edge(stream, kds, src, pe.a, ke, E);
      gemm_edge(stream, vds, src, pe.m, ve, E);
      k_edge_score<<<E / 8, 256, 0, stream>>>(qdd, dst, ke, pe.p, al, E);
      hipMemsetAsync(amax, 0, (size_t)Nd * 4, stream);   // key 0 decodes to non-finite
      hipMemsetAsync(den, 0, (size_t)Nd * 4, stream);
      k_amax<<<(E + 255) / 256, 256, 0, stream>>>(al, dst, amax, E);
      k_expden<<<(E + 255) / 256, 256, 0, stream>>>(al, dst, amax, ex, den, E);
      k_scatter<<<(unsigned)(((size_t)E * ED_DIM + 255) / 256), 256, 0, stream>>>(ex, den, dst, ve, aggd, E);
    };
    rel(P.follow, kdu, vdu, qdu, e_fol, e_fol + NE_FF, NE_FF, U_CNT, aggu);
    rel(P.frnd,   kdu, vdu, qdu, e_fri, e_fri + NE_FF, NE_FF, U_CNT, aggu);
    rel(P.post,   kdu, vdu, qdt, e_pst, e_pst + NE_P,  NE_P,  NT_CNT, aggt);
    rel(P.rev,    kdt, vdt, qdu, e_rev, e_rev + NE_P,  NE_P,  U_CNT, aggu);

    k_gelu<<<(unsigned)(((size_t)U_CNT * ED_DIM + 255) / 256), 256, 0, stream>>>(aggu, (size_t)U_CNT * ED_DIM);
    k_gelu<<<(unsigned)(((size_t)NT_CNT * ED_DIM + 255) / 256), 256, 0, stream>>>(aggt, (size_t)NT_CNT * ED_DIM);
    // reuse ke/ve as the post-aggregation linear outputs
    gemm_nn(stream, aggu, ED_DIM, P.nu.Wa, ED_DIM, P.nu.ba, ke, ED_DIM, U_CNT, ED_DIM, ED_DIM, 0);
    gemm_nn(stream, aggt, ED_DIM, P.nt.Wa, ED_DIM, P.nt.ba, ve, ED_DIM, NT_CNT, ED_DIM, ED_DIM, 0);
    k_blend<<<(unsigned)(((size_t)U_CNT * ED_DIM + 255) / 256), 256, 0, stream>>>(ke, xui, P.nu.skip, xuo, (size_t)U_CNT * ED_DIM);
    k_blend<<<(unsigned)(((size_t)NT_CNT * ED_DIM + 255) / 256), 256, 0, stream>>>(ve, xti, P.nt.skip, xto, (size_t)NT_CNT * ED_DIM);
  };

  runHgt(H1, xu, xt, xu1, xt1);
  runHgt(H2, xu1, xt1, xu2, xt2);

  // =========== Phase 4: classifier ===========
  // (kdu buffer is free now — reuse for the hidden layer)
  gemm_nn(stream, xu2, ED_DIM, cW1, ED_DIM, cb1, kdu, ED_DIM, U_CNT, ED_DIM, ED_DIM, 1);
  k_cls<<<(U_CNT + 255) / 256, 256, 0, stream>>>(kdu, cW2, cb2, (float*)d_out, U_CNT);
}
